// generator_19894288515260
// MI455X (gfx1250) — matrix operand order repaired and checked by reading, not compiled, not run on hardware
//
#include <hip/hip_runtime.h>
#include <math.h>

// ---------------- sizes ----------------
#define B_    100
#define T_    30
#define U_    256
#define ED_   100
#define V_    32000   // HVS == MVS
#define MP_   112     // M padded to WMMA tile multiple (7 * 16)
#define RS_   112     // result row stride (padded)

typedef __attribute__((ext_vector_type(16))) __bf16         v16bf;
typedef __attribute__((ext_vector_type(8)))  float          v8f;
typedef __attribute__((ext_vector_type(8)))  unsigned short ushort8;

union ABfrag { v16bf v; ushort8 h[2]; };

static __device__ __forceinline__ unsigned short f2bf(float f) {
    unsigned int u = __float_as_uint(f);
    unsigned int r = u + 0x7FFFu + ((u >> 16) & 1u);   // round-to-nearest-even
    return (unsigned short)(r >> 16);
}
static __device__ __forceinline__ ushort8 ld8(const unsigned short* p) {
    return *(const ushort8*)p;
}
static __device__ __forceinline__ float sigf(float x) { return 1.f / (1.f + expf(-x)); }

// ---------------- prep: bf16 transposes + zero pad rows (run once per call) -------
__global__ void prep_woutT(const float* __restrict__ Wout, unsigned short* __restrict__ WoutT) {
    size_t idx = (size_t)blockIdx.x * blockDim.x + threadIdx.x;   // over 256*32000
    if (idx >= (size_t)U_ * V_) return;
    int k = (int)(idx / V_), v = (int)(idx % V_);
    WoutT[(size_t)v * U_ + k] = f2bf(Wout[idx]);                  // [V,K] K-contiguous
}
__global__ void prep_emT(const float* __restrict__ Em, unsigned short* __restrict__ EmT) {
    size_t idx = (size_t)blockIdx.x * blockDim.x + threadIdx.x;   // over 32000*100
    if (idx >= (size_t)V_ * ED_) return;
    int v = (int)(idx / ED_), e = (int)(idx % ED_);
    EmT[(size_t)e * V_ + v] = f2bf(Em[idx]);                      // [Epad, V] K-contiguous
}
// zero pad rows [100,112) of obf and EmT once; they are never written again
__global__ void zero_pads(unsigned short* __restrict__ obf, unsigned short* __restrict__ EmT) {
    size_t idx = (size_t)blockIdx.x * blockDim.x + threadIdx.x;   // over 12*32000
    if (idx >= (size_t)(MP_ - B_) * V_) return;
    obf[(size_t)B_ * V_ + idx] = 0;
    EmT[(size_t)ED_ * V_ + idx] = 0;
}

// ---------------- embedding gather ----------------
__global__ void embed_k(const int* __restrict__ tok, const float* __restrict__ Eh,
                        float* __restrict__ x) {
    int idx = blockIdx.x * blockDim.x + threadIdx.x;              // B*T*ED
    if (idx >= B_ * T_ * ED_) return;
    int e = idx % ED_, bt = idx / ED_;
    x[idx] = Eh[(size_t)tok[bt] * ED_ + e];
}

// ---------------- zero/initial state (attb zeroed over full padded 112 rows) ------
__global__ void init_state(float* h0, float* c, float* alpha, float* att,
                           unsigned short* attb, float* nxt0) {
    int idx = blockIdx.x * blockDim.x + threadIdx.x;
    if (idx < B_ * U_) { h0[idx] = 0.f; c[idx] = 0.f; att[idx] = 0.f; nxt0[idx] = 0.f; }
    if (idx < MP_ * U_) attb[idx] = 0;
    if (idx < B_ * T_) alpha[idx] = ((idx % T_) == 0) ? 1.f : 0.f;
}
__global__ void zero_result(float* r) {
    int i = blockIdx.x * blockDim.x + threadIdx.x;
    if (i < MP_ * RS_) r[i] = 0.f;
}

// ---------------- encoder LSTM cell (f32 VALU, one thread per (b,u)) ----------------
__global__ void enc_cell(const float* __restrict__ x, const float* __restrict__ Wk,
                         const float* __restrict__ Wr, const float* __restrict__ be,
                         const float* __restrict__ h_in, float* __restrict__ h_out,
                         float* __restrict__ c, float* __restrict__ memory, int t) {
    int idx = blockIdx.x * blockDim.x + threadIdx.x;
    if (idx >= B_ * U_) return;
    int b = idx >> 8, u = idx & 255;
    float zi = be[u], zf = be[U_ + u], zg = be[2 * U_ + u], zo = be[3 * U_ + u];
    const float* xb = x + (b * T_ + t) * ED_;
    for (int k = 0; k < ED_; ++k) {
        float xv = xb[k]; const float* w = Wk + k * 4 * U_;
        zi += xv * w[u]; zf += xv * w[U_ + u]; zg += xv * w[2 * U_ + u]; zo += xv * w[3 * U_ + u];
    }
    const float* hb = h_in + b * U_;
    for (int k = 0; k < U_; ++k) {
        float hv = hb[k]; const float* w = Wr + k * 4 * U_;
        zi += hv * w[u]; zf += hv * w[U_ + u]; zg += hv * w[2 * U_ + u]; zo += hv * w[3 * U_ + u];
    }
    float cc = c[idx];
    cc = sigf(zf) * cc + sigf(zi) * tanhf(zg);
    float hh = sigf(zo) * tanhf(cc);
    c[idx] = cc;
    h_out[idx] = hh;
    memory[(b * T_ + t) * U_ + u] = hh;
}

// ---------------- keys = memory @ Wmem ----------------
__global__ void keys_gemm(const float* __restrict__ memory, const float* __restrict__ Wmem,
                          float* __restrict__ keys) {
    int idx = blockIdx.x * blockDim.x + threadIdx.x;              // B*T*U
    if (idx >= B_ * T_ * U_) return;
    int u = idx & 255, bt = idx >> 8;
    const float* m = memory + bt * U_;
    float s = 0.f;
    for (int k = 0; k < U_; ++k) s += m[k] * Wmem[k * U_ + u];
    keys[idx] = s;
}

// ---------------- decoder LSTM cell: cell_in = [nxt, att] ----------------
__global__ void dec_cell(const float* __restrict__ nxt, const float* __restrict__ att,
                         const float* __restrict__ Wk, const float* __restrict__ Wr,
                         const float* __restrict__ bd, const float* __restrict__ h_in,
                         float* __restrict__ h_out, float* __restrict__ c) {
    int idx = blockIdx.x * blockDim.x + threadIdx.x;
    if (idx >= B_ * U_) return;
    int b = idx >> 8, u = idx & 255;
    float zi = bd[u], zf = bd[U_ + u], zg = bd[2 * U_ + u], zo = bd[3 * U_ + u];
    const float* nb = nxt + b * U_;
    for (int k = 0; k < U_; ++k) {
        float v = nb[k]; const float* w = Wk + k * 4 * U_;
        zi += v * w[u]; zf += v * w[U_ + u]; zg += v * w[2 * U_ + u]; zo += v * w[3 * U_ + u];
    }
    const float* ab = att + b * U_;
    for (int k = 0; k < U_; ++k) {
        float v = ab[k]; const float* w = Wk + (U_ + k) * 4 * U_;
        zi += v * w[u]; zf += v * w[U_ + u]; zg += v * w[2 * U_ + u]; zo += v * w[3 * U_ + u];
    }
    const float* hb = h_in + b * U_;
    for (int k = 0; k < U_; ++k) {
        float v = hb[k]; const float* w = Wr + k * 4 * U_;
        zi += v * w[u]; zf += v * w[U_ + u]; zg += v * w[2 * U_ + u]; zo += v * w[3 * U_ + u];
    }
    float cc = c[idx];
    cc = sigf(zf) * cc + sigf(zi) * tanhf(zg);
    float hh = sigf(zo) * tanhf(cc);
    c[idx] = cc;
    h_out[idx] = hh;
}

// ---------------- monotonic attention (one block per batch row) ----------------
__global__ void attention_k(const float* __restrict__ h, const float* __restrict__ keys,
                            const float* __restrict__ memory, float* __restrict__ alpha,
                            float* __restrict__ att, unsigned short* __restrict__ attb) {
    __shared__ float sh[U_];
    __shared__ float sp[T_];
    __shared__ float sal[T_];
    int b = blockIdx.x, tid = threadIdx.x;
    sh[tid] = h[b * U_ + tid];
    __syncthreads();
    if (tid < T_) {
        const float* kr = keys + (b * T_ + tid) * U_;
        float s = 0.f;
        for (int k = 0; k < U_; ++k) s += sh[k] * kr[k];
        sp[tid] = sigf(s);
    }
    __syncthreads();
    if (tid == 0) {  // sequential scan: cumprod / cumsum of monotonic attention
        float run_cp = 1.f, run_sum = 0.f;
        for (int j = 0; j < T_; ++j) {
            float pj = sp[j];
            float cpe = run_cp;                               // prod of clipped (1-p) over [0, j)
            float q = fmaxf(fminf(1.f - pj, 1.f), 1e-10f);
            run_cp *= q;
            run_sum += alpha[b * T_ + j] / fmaxf(cpe, 1e-10f);
            sal[j] = pj * cpe * run_sum;
        }
    }
    __syncthreads();
    if (tid < T_) alpha[b * T_ + tid] = sal[tid];
    float a = 0.f;
    for (int j = 0; j < T_; ++j) a += sal[j] * memory[(b * T_ + j) * U_ + tid];
    att[b * U_ + tid] = a;
    attb[b * U_ + tid] = f2bf(a);      // rows [100,112) stay zero from init
}

// ---------------- logits = att @ Wout + bout  (bf16 WMMA, dominant GEMM) ----------------
// One wave per (M-tile, 4 consecutive N-tiles). M=112 (padded), N=32000, K=256.
// A fragment loaded once per K-step, reused across 4 independent WMMAs (no guards:
// all buffers physically padded/zeroed, so the K loop is pure b128 loads + wmma).
// A frag: row M = lane%16; 16 K-contiguous bf16 at k0 + (lane>=16 ? 16 : 0).
// B frag: col N = lane%16; 16 K-contiguous bf16 at k0 + (lane>=16 ? 16 : 0).
// C/D:    col N = lane%16; row M = vgpr + 8*(lane>=16).
__global__ void logits_wmma(const unsigned short* __restrict__ attb,
                            const unsigned short* __restrict__ WoutT,
                            const float* __restrict__ bout, float* __restrict__ S) {
    int wave = (blockIdx.x * blockDim.x + threadIdx.x) >> 5;
    if (wave >= 7 * 500) return;                      // whole-wave guard (EXEC stays all-1s)
    int lane = threadIdx.x & 31;
    int l15 = lane & 15, lh = lane >> 4;
    int mt = wave % 7, ntg = wave / 7;                // ntg covers 4 N-tiles
    int m  = mt * 16 + l15;
    int v0 = ntg * 64 + l15;                          // first of 4 tile columns for this lane
    v8f acc0 = {0.f,0.f,0.f,0.f,0.f,0.f,0.f,0.f};
    v8f acc1 = acc0, acc2 = acc0, acc3 = acc0;
    const unsigned short* ap = attb + (size_t)m * U_ + lh * 16;
    const unsigned short* b0 = WoutT + (size_t)(v0     ) * U_ + lh * 16;
    const unsigned short* b1 = WoutT + (size_t)(v0 + 16) * U_ + lh * 16;
    const unsigned short* b2 = WoutT + (size_t)(v0 + 32) * U_ + lh * 16;
    const unsigned short* b3 = WoutT + (size_t)(v0 + 48) * U_ + lh * 16;
    #pragma unroll
    for (int k0 = 0; k0 < U_; k0 += 32) {
        ABfrag A, Bf0, Bf1, Bf2, Bf3;
        A.h[0]   = ld8(ap + k0);     A.h[1]   = ld8(ap + k0 + 8);
        Bf0.h[0] = ld8(b0 + k0);     Bf0.h[1] = ld8(b0 + k0 + 8);
        Bf1.h[0] = ld8(b1 + k0);     Bf1.h[1] = ld8(b1 + k0 + 8);
        Bf2.h[0] = ld8(b2 + k0);     Bf2.h[1] = ld8(b2 + k0 + 8);
        Bf3.h[0] = ld8(b3 + k0);     Bf3.h[1] = ld8(b3 + k0 + 8);
        acc0 = __builtin_amdgcn_wmma_f32_16x16x32_bf16(false, A.v, false, Bf0.v, (short)0, acc0, false, false);
        acc1 = __builtin_amdgcn_wmma_f32_16x16x32_bf16(false, A.v, false, Bf1.v, (short)0, acc1, false, false);
        acc2 = __builtin_amdgcn_wmma_f32_16x16x32_bf16(false, A.v, false, Bf2.v, (short)0, acc2, false, false);
        acc3 = __builtin_amdgcn_wmma_f32_16x16x32_bf16(false, A.v, false, Bf3.v, (short)0, acc3, false, false);
    }
    int mbase = mt * 16 + lh * 8;                     // S is padded to 112 rows: no store guard
    #pragma unroll
    for (int j = 0; j < 4; ++j) {
        int v = v0 + j * 16;
        float bo = bout[v];
        v8f* a = (j == 0) ? &acc0 : (j == 1) ? &acc1 : (j == 2) ? &acc2 : &acc3;
        #pragma unroll
        for (int i = 0; i < 8; ++i)
            S[(size_t)(mbase + i) * V_ + v] = (*a)[i] + bo;
    }
}

// ---------------- softmax + argmax over V=32000 (one block per b) ----------------
__global__ void softmax_o(const float* __restrict__ S, unsigned short* __restrict__ obf,
                          float* __restrict__ r_out, int t) {
    __shared__ float smax[256];
    __shared__ int   sidx[256];
    __shared__ float ssum[256];
    int b = blockIdx.x, tid = threadIdx.x;
    const float* row = S + (size_t)b * V_;
    float mx = -3.402823466e38f; int mi = 0;
    for (int v = tid; v < V_; v += 256) {
        float val = row[v];
        if (val > mx) { mx = val; mi = v; }
    }
    smax[tid] = mx; sidx[tid] = mi;
    __syncthreads();
    for (int s = 128; s > 0; s >>= 1) {
        if (tid < s) {
            float o = smax[tid + s]; int oi = sidx[tid + s];
            if (o > smax[tid] || (o == smax[tid] && oi < sidx[tid])) { smax[tid] = o; sidx[tid] = oi; }
        }
        __syncthreads();
    }
    float gmax = smax[0];
    float sum = 0.f;
    for (int v = tid; v < V_; v += 256) sum += expf(row[v] - gmax);
    ssum[tid] = sum;
    __syncthreads();
    for (int s = 128; s > 0; s >>= 1) {
        if (tid < s) ssum[tid] += ssum[tid + s];
        __syncthreads();
    }
    float inv = 1.f / ssum[0];
    for (int v = tid; v < V_; v += 256)
        obf[(size_t)b * V_ + v] = f2bf(expf(row[v] - gmax) * inv);
    if (tid == 0) r_out[b * T_ + t] = (float)sidx[0];   // argmax(softmax) == argmax(logits)
}

// ---------------- result = o @ E_machine (bf16 WMMA, K split 10 ways + f32 atomics) -----
// All operands padded to 112 rows -> no guards anywhere; pad outputs are exact zeros.
__global__ void result_wmma(const unsigned short* __restrict__ obf,
                            const unsigned short* __restrict__ EmT,
                            float* __restrict__ result) {
    int wave = (blockIdx.x * blockDim.x + threadIdx.x) >> 5;
    if (wave >= 7 * 7 * 10) return;                   // whole-wave guard
    int lane = threadIdx.x & 31;
    int l15 = lane & 15, lh = lane >> 4;
    int mt = wave % 7, nt = (wave / 7) % 7, kc = wave / 49;
    int m = mt * 16 + l15;                            // b row (padded)
    int e = nt * 16 + l15;                            // out col (padded)
    v8f acc = {0.f,0.f,0.f,0.f,0.f,0.f,0.f,0.f};
    const unsigned short* ap = obf + (size_t)m * V_ + lh * 16;
    const unsigned short* bp = EmT + (size_t)e * V_ + lh * 16;
    int kend = (kc + 1) * 3200;
    for (int k0 = kc * 3200; k0 < kend; k0 += 32) {
        ABfrag A, Bf;
        A.h[0]  = ld8(ap + k0);  A.h[1]  = ld8(ap + k0 + 8);
        Bf.h[0] = ld8(bp + k0);  Bf.h[1] = ld8(bp + k0 + 8);
        acc = __builtin_amdgcn_wmma_f32_16x16x32_bf16(false, A.v, false, Bf.v,
                                                      (short)0, acc, false, false);
    }
    int mbase = mt * 16 + lh * 8;
    #pragma unroll
    for (int i = 0; i < 8; ++i)
        atomicAdd(&result[(mbase + i) * RS_ + e], acc[i]);
}

// ---------------- passable[:, t] = result.mean(axis=0) ----------------
__global__ void mean_out(const float* __restrict__ result, float* __restrict__ passable, int t) {
    int e = blockIdx.x * blockDim.x + threadIdx.x;
    if (e >= ED_) return;
    float s = 0.f;
    for (int b = 0; b < B_; ++b) s += result[b * RS_ + e];
    passable[e * T_ + t] = s * (1.f / (float)B_);
}

// ---------------- nxt = concat(result, nxt) @ Wstep[t] + bstep[t] ----------------
__global__ void nxt_update(const float* __restrict__ result, const float* __restrict__ nxt_in,
                           const float* __restrict__ Wstep, const float* __restrict__ bstep,
                           float* __restrict__ nxt_out, int t) {
    int idx = blockIdx.x * blockDim.x + threadIdx.x;
    if (idx >= B_ * U_) return;
    int b = idx >> 8, u = idx & 255;
    const float* Ws = Wstep + (size_t)t * (ED_ + U_) * U_;
    float acc = bstep[t * U_ + u];
    for (int k = 0; k < ED_; ++k) acc += result[b * RS_ + k] * Ws[k * U_ + u];
    const float* nb = nxt_in + b * U_;
    for (int k = 0; k < U_; ++k) acc += nb[k] * Ws[(ED_ + k) * U_ + u];
    nxt_out[idx] = acc;
}

// ======================= host-side launcher =======================
extern "C" void kernel_launch(void* const* d_in, const int* in_sizes, int n_in,
                              void* d_out, int out_size, void* d_ws, size_t ws_size,
                              hipStream_t stream) {
    const int*   tokens   = (const int*)d_in[0];
    const float* E_human  = (const float*)d_in[1];
    const float* E_mach   = (const float*)d_in[2];
    const float* We_k     = (const float*)d_in[3];
    const float* We_r     = (const float*)d_in[4];
    const float* be       = (const float*)d_in[5];
    const float* Wd_k     = (const float*)d_in[6];
    const float* Wd_r     = (const float*)d_in[7];
    const float* bd       = (const float*)d_in[8];
    const float* Wmem     = (const float*)d_in[9];
    const float* Wout     = (const float*)d_in[10];
    const float* bout     = (const float*)d_in[11];
    const float* Wstep    = (const float*)d_in[12];
    const float* bstep    = (const float*)d_in[13];

    float* out_f = (float*)d_out;          // [0,3000): passable [ED,T]; [3000,6000): r [B,T]

    // workspace sub-allocation (256B aligned)
    char* ws = (char*)d_ws;
    size_t off = 0;
    auto alloc = [&](size_t bytes) -> void* {
        void* p = ws + off;
        off = (off + bytes + 255) & ~(size_t)255;
        return p;
    };
    float*          x      = (float*)alloc((size_t)B_ * T_ * ED_ * 4);
    float*          memory = (float*)alloc((size_t)B_ * T_ * U_ * 4);
    float*          keys   = (float*)alloc((size_t)B_ * T_ * U_ * 4);
    float*          h0     = (float*)alloc((size_t)B_ * U_ * 4);
    float*          h1     = (float*)alloc((size_t)B_ * U_ * 4);
    float*          cbuf   = (float*)alloc((size_t)B_ * U_ * 4);
    float*          alpha  = (float*)alloc((size_t)B_ * T_ * 4);
    float*          att    = (float*)alloc((size_t)B_ * U_ * 4);
    unsigned short* attb   = (unsigned short*)alloc((size_t)MP_ * U_ * 2);   // padded
    float*          nxt0   = (float*)alloc((size_t)B_ * U_ * 4);
    float*          nxt1   = (float*)alloc((size_t)B_ * U_ * 4);
    float*          S      = (float*)alloc((size_t)MP_ * V_ * 4);            // padded
    unsigned short* obf    = (unsigned short*)alloc((size_t)MP_ * V_ * 2);   // padded
    float*          result = (float*)alloc((size_t)MP_ * RS_ * 4);           // padded
    unsigned short* WoutT  = (unsigned short*)alloc((size_t)U_ * V_ * 2);
    unsigned short* EmT    = (unsigned short*)alloc((size_t)MP_ * V_ * 2);   // padded
    (void)in_sizes; (void)n_in; (void)out_size; (void)ws_size;

    float* hbuf[2] = {h0, h1};
    float* nbuf[2] = {nxt0, nxt1};

    // prep: bf16 transposed weights (Wout: 16MB, Em: ~7MB — L2-resident afterwards)
    prep_woutT<<<((size_t)U_ * V_ + 255) / 256, 256, 0, stream>>>(Wout, WoutT);
    prep_emT<<<((size_t)V_ * ED_ + 255) / 256, 256, 0, stream>>>(E_mach, EmT);
    zero_pads<<<(((size_t)(MP_ - B_) * V_) + 255) / 256, 256, 0, stream>>>(obf, EmT);
    embed_k<<<(B_ * T_ * ED_ + 255) / 256, 256, 0, stream>>>(tokens, E_human, x);
    init_state<<<(MP_ * U_ + 255) / 256, 256, 0, stream>>>(h0, cbuf, alpha, att, attb, nxt0);

    // encoder (sequential over T); h ping-pong, final h lands in hbuf[0]
    for (int t = 0; t < T_; ++t)
        enc_cell<<<(B_ * U_ + 255) / 256, 256, 0, stream>>>(
            x, We_k, We_r, be, hbuf[t & 1], hbuf[1 - (t & 1)], cbuf, memory, t);

    keys_gemm<<<(B_ * T_ * U_ + 255) / 256, 256, 0, stream>>>(memory, Wmem, keys);

    // decoder (sequential over T); h/c continue from encoder state
    for (int t = 0; t < T_; ++t) {
        const float* h_in  = hbuf[t & 1];
        float*       h_out = hbuf[1 - (t & 1)];
        dec_cell<<<(B_ * U_ + 255) / 256, 256, 0, stream>>>(
            nbuf[t & 1], att, Wd_k, Wd_r, bd, h_in, h_out, cbuf);
        attention_k<<<B_, 256, 0, stream>>>(h_out, keys, memory, alpha, att, attb);
        // 7 M-tiles x 500 groups-of-4-N-tiles = 3500 waves -> 438 blocks of 8 waves
        logits_wmma<<<438, 256, 0, stream>>>(attb, WoutT, bout, S);
        softmax_o<<<B_, 256, 0, stream>>>(S, obf, out_f + 3000, t);
        zero_result<<<(MP_ * RS_ + 255) / 256, 256, 0, stream>>>(result);
        // 7x7 tiles x 10 K-chunks = 490 waves -> 62 blocks (last waves guarded)
        result_wmma<<<62, 256, 0, stream>>>(obf, EmT, result);
        mean_out<<<1, 128, 0, stream>>>(result, out_f, t);
        nxt_update<<<(B_ * U_ + 255) / 256, 256, 0, stream>>>(
            result, nbuf[t & 1], Wstep, bstep, nbuf[1 - (t & 1)], t);
    }
}
